// VectorQuantizer_26603027431781
// MI455X (gfx1250) — compile-verified, matrix operand order checked by reading, not run on hardware
//
#include <hip/hip_runtime.h>
#include <hip/hip_bf16.h>
#include <stdint.h>

// ---------------------------------------------------------------------------
// VQ codebook lookup on MI455X (gfx1250, wave32, WMMA).
//   scores = ||w||^2 - 2 * Z @ W^T   (argmin per row), z_q = W[argmin]
// GEMM: V_WMMA_F32_16X16X32_BF16 with 3-term split precision
// (zh*wh + zl*wh + zh*wl) ~ fp32 accuracy at bf16 WMMA rate.
// Wave tiling 32x64 (MT=2 x NT=4) to reuse LDS A-frags 4x (LDS-bw limit).
// z tile staged via GLOBAL_LOAD_ASYNC_TO_LDS_B128 (ASYNCcnt path).
// ---------------------------------------------------------------------------

typedef __bf16 bf16_t;
typedef __attribute__((ext_vector_type(16))) __bf16 v16bf;
typedef __attribute__((ext_vector_type(8)))  float  v8f;

// async-to-LDS builtin operand types: v4i in addrspace(1)/(3)
typedef int v4i __attribute__((vector_size(16)));
typedef __attribute__((address_space(1))) v4i* global_v4i_ptr;
typedef __attribute__((address_space(3))) v4i* lds_v4i_ptr;

#define KCODES 8192
#define EDIM   256
#define NPTS   32768
#define MTILE  128          // z rows per block (bf16 hi/lo tile in LDS)
#define CTILE  128          // codes per iteration (2 N-groups x 64)
#define MT     2            // 16-row tiles per wave
#define NT     4            // 16-code tiles per wave

#define LDS_ZH   0
#define LDS_ZL   (MTILE * EDIM * 2)            // 65536
#define LDS_BEST (2 * MTILE * EDIM * 2)        // 131072
#define LDS_RAW  (LDS_BEST + MTILE * 8)        // 132096
#define LDS_TOT  (LDS_RAW + 8 * 2048)          // 148480 (2 blocks / WGP)

#if defined(__HIP_DEVICE_COMPILE__) && __has_builtin(__builtin_amdgcn_global_load_async_to_lds_b128)
#define HAVE_ASYNC_LDS 1
#else
#define HAVE_ASYNC_LDS 0
#endif

#if defined(__HIP_DEVICE_COMPILE__) && __has_builtin(__builtin_amdgcn_s_wait_asynccnt)
#define WAIT_ASYNC() __builtin_amdgcn_s_wait_asynccnt(0)
#else
#define WAIT_ASYNC() asm volatile("s_wait_asynccnt 0x0" ::: "memory")
#endif

union Frag { v16bf v; uint4 q[2]; };

__device__ __forceinline__ uint32_t f32_ord(float f) {
    // order-preserving map: smaller float -> smaller u32
    uint32_t b = __float_as_uint(f);
    return b ^ ((b & 0x80000000u) ? 0xFFFFFFFFu : 0x80000000u);
}

__device__ __forceinline__ unsigned long long shflx64(unsigned long long v, int m) {
    unsigned lo = (unsigned)v, hi = (unsigned)(v >> 32);
    lo = (unsigned)__shfl_xor((int)lo, m, 32);
    hi = (unsigned)__shfl_xor((int)hi, m, 32);
    return ((unsigned long long)hi << 32) | lo;
}

// --- prep: W (fp32) -> Wh/Wl (bf16 hi/lo) + exact fp32 ||w||^2 --------------
__global__ __launch_bounds__(256) void vq_prep(const float* __restrict__ W,
                                               bf16_t* __restrict__ Wh,
                                               bf16_t* __restrict__ Wl,
                                               float*  __restrict__ wnorm) {
    const int k = blockIdx.x;
    const int d = threadIdx.x;
    float w = W[(long long)k * EDIM + d];
    bf16_t h = (bf16_t)w;
    bf16_t l = (bf16_t)(w - (float)h);
    Wh[(long long)k * EDIM + d] = h;
    Wl[(long long)k * EDIM + d] = l;

    float s = w * w;
#pragma unroll
    for (int m = 16; m >= 1; m >>= 1) s += __shfl_xor(s, m, 32);
    __shared__ float part[8];
    if ((threadIdx.x & 31) == 0) part[threadIdx.x >> 5] = s;
    __syncthreads();
    if (threadIdx.x == 0) {
        float t = 0.f;
#pragma unroll
        for (int i = 0; i < 8; ++i) t += part[i];
        wnorm[k] = t;
    }
}

// --- main: fused GEMM + argmin + gather -------------------------------------
__global__ __launch_bounds__(256) void vq_main(const float* __restrict__ z,
                                               const float* __restrict__ W,
                                               const bf16_t* __restrict__ Wh,
                                               const bf16_t* __restrict__ Wl,
                                               const float* __restrict__ wnorm,
                                               float* __restrict__ out_zq,
                                               float* __restrict__ out_idx,
                                               float* __restrict__ out_loss) {
    extern __shared__ char smem[];
    bf16_t* zh = (bf16_t*)(smem + LDS_ZH);
    bf16_t* zl = (bf16_t*)(smem + LDS_ZL);
    unsigned long long* best = (unsigned long long*)(smem + LDS_BEST);

    const int tid  = threadIdx.x;
    const int wv   = tid >> 5;
    const int mg   = wv & 3;            // M-group: rows mg*32 .. mg*32+31
    const int ng   = wv >> 2;           // N-group: codes ng*64 .. ng*64+63
    const int lane = tid & 31;
    const int nl   = lane & 15;
    const int hk   = lane >> 4;         // half: K-offset +8, M-offset +8
    const int koff = hk * 8;
    const long long rowBase = (long long)blockIdx.x * MTILE;

    // ---- stage z tile -> bf16 hi/lo in LDS (fp32 split for 3xBF16) --------
#if HAVE_ASYNC_LDS
    {
        // each wave DMAs its 16 rows through a private 2KB LDS scratch chunk
        float* rawf = (float*)(smem + LDS_RAW) + wv * 512;
        for (int c = 0; c < 8; ++c) {                    // 2 rows per chunk
            const char* gsrc =
                (const char*)(z + (rowBase + wv * 16 + c * 2) * EDIM);
#pragma unroll
            for (int j = 0; j < 4; ++j) {
                int off = j * 512 + lane * 16;
                __builtin_amdgcn_global_load_async_to_lds_b128(
                    (global_v4i_ptr)(uintptr_t)(gsrc + off),
                    (lds_v4i_ptr)(unsigned)(uintptr_t)((char*)rawf + off),
                    0, 0);
            }
            WAIT_ASYNC();
            int ebase = (wv * 16 + c * 2) * EDIM + lane * 16;
#pragma unroll
            for (int t = 0; t < 4; ++t) {
                float4 v = *(const float4*)(rawf + lane * 16 + t * 4);
                float f0 = v.x, f1 = v.y, f2 = v.z, f3 = v.w;
                bf16_t h0 = (bf16_t)f0, h1 = (bf16_t)f1,
                       h2 = (bf16_t)f2, h3 = (bf16_t)f3;
                int e = ebase + t * 4;
                zh[e] = h0; zh[e + 1] = h1; zh[e + 2] = h2; zh[e + 3] = h3;
                zl[e]     = (bf16_t)(f0 - (float)h0);
                zl[e + 1] = (bf16_t)(f1 - (float)h1);
                zl[e + 2] = (bf16_t)(f2 - (float)h2);
                zl[e + 3] = (bf16_t)(f3 - (float)h3);
            }
        }
    }
#else
    for (int i = tid; i < MTILE * EDIM / 4; i += 256) {
        float4 v = ((const float4*)(z + rowBase * EDIM))[i];
        int e = i * 4;
        float f0 = v.x, f1 = v.y, f2 = v.z, f3 = v.w;
        bf16_t h0 = (bf16_t)f0, h1 = (bf16_t)f1, h2 = (bf16_t)f2, h3 = (bf16_t)f3;
        zh[e] = h0; zh[e + 1] = h1; zh[e + 2] = h2; zh[e + 3] = h3;
        zl[e]     = (bf16_t)(f0 - (float)h0);
        zl[e + 1] = (bf16_t)(f1 - (float)h1);
        zl[e + 2] = (bf16_t)(f2 - (float)h2);
        zl[e + 3] = (bf16_t)(f3 - (float)h3);
    }
#endif
    if (tid < MTILE) best[tid] = ~0ULL;
    __syncthreads();

    for (int ct = 0; ct < KCODES / CTILE; ++ct) {
        int   cr[NT];
        float wn[NT];
#pragma unroll
        for (int nt = 0; nt < NT; ++nt) {
            cr[nt] = ct * CTILE + ng * 64 + nt * 16 + nl;   // lane's code column
            wn[nt] = wnorm[cr[nt]];
        }

        if (ct + 1 < KCODES / CTILE) {      // global_prefetch next W strips
            __builtin_prefetch(Wh + (long long)(cr[0] + CTILE) * EDIM, 0, 1);
            __builtin_prefetch(Wl + (long long)(cr[2] + CTILE) * EDIM, 0, 1);
        }

        v8f acc[MT][NT];
#pragma unroll
        for (int mt = 0; mt < MT; ++mt)
#pragma unroll
            for (int nt = 0; nt < NT; ++nt)
                acc[mt][nt] = v8f{0.f, 0.f, 0.f, 0.f, 0.f, 0.f, 0.f, 0.f};

#pragma unroll
        for (int kb = 0; kb < 8; ++kb) {            // D = 8 x 32
            Frag Ah[MT], Al[MT];                    // A-frags reused across NT
#pragma unroll
            for (int mt = 0; mt < MT; ++mt) {
                const int ro = (mg * 32 + mt * 16 + nl) * EDIM + kb * 32 + koff;
                Ah[mt].q[0] = *(const uint4*)(zh + ro);
                Ah[mt].q[1] = *(const uint4*)(zh + ro + 16);
                Al[mt].q[0] = *(const uint4*)(zl + ro);
                Al[mt].q[1] = *(const uint4*)(zl + ro + 16);
            }
#pragma unroll
            for (int nt = 0; nt < NT; ++nt) {
                const bf16_t* wb  = Wh + (long long)cr[nt] * EDIM + kb * 32 + koff;
                const bf16_t* wlb = Wl + (long long)cr[nt] * EDIM + kb * 32 + koff;
                Frag Bh, Bl;                        // B-frag: lane=column (W row)
                Bh.q[0] = *(const uint4*)(wb);
                Bh.q[1] = *(const uint4*)(wb + 16);
                Bl.q[0] = *(const uint4*)(wlb);
                Bl.q[1] = *(const uint4*)(wlb + 16);
#pragma unroll
                for (int mt = 0; mt < MT; ++mt) {
                    acc[mt][nt] = __builtin_amdgcn_wmma_f32_16x16x32_bf16(
                        false, Ah[mt].v, false, Bh.v, (short)0, acc[mt][nt], false, false);
                    acc[mt][nt] = __builtin_amdgcn_wmma_f32_16x16x32_bf16(
                        false, Al[mt].v, false, Bh.v, (short)0, acc[mt][nt], false, false);
                    acc[mt][nt] = __builtin_amdgcn_wmma_f32_16x16x32_bf16(
                        false, Ah[mt].v, false, Bl.v, (short)0, acc[mt][nt], false, false);
                }
            }
        }

        // score, pack (ord(score)<<32 | code), butterfly min, LDS ds_min_u64
#pragma unroll
        for (int mt = 0; mt < MT; ++mt) {
#pragma unroll
            for (int nt = 0; nt < NT; ++nt) {
#pragma unroll
                for (int v = 0; v < 8; ++v) {
                    float s = wn[nt] - 2.0f * acc[mt][nt][v];
                    unsigned long long key =
                        ((unsigned long long)f32_ord(s) << 32) | (unsigned)cr[nt];
#pragma unroll
                    for (int m = 1; m < 16; m <<= 1) {
                        unsigned long long o = shflx64(key, m);
                        key = (o < key) ? o : key;
                    }
                    if (nl == v)
                        atomicMin(&best[mg * 32 + mt * 16 + v + hk * 8], key);
                }
            }
        }
    }
    __syncthreads();

    // Decode argmin, write idx (as float) + zero loss, gather fp32 codebook row.
    {
        const int r = tid >> 1, half = tid & 1;
        unsigned long long key = best[r];
        unsigned idx = (unsigned)(key & 0xFFFFFFFFu);
        long long g = rowBase + r;
        if (half == 0) { out_idx[g] = (float)idx; out_loss[g] = 0.0f; }
        const float4* src = (const float4*)(W + (long long)idx * EDIM) + half * 32;
        float4* dst = (float4*)(out_zq + g * EDIM) + half * 32;
#pragma unroll
        for (int i = 0; i < 32; ++i) dst[i] = src[i];
    }
}

extern "C" void kernel_launch(void* const* d_in, const int* in_sizes, int n_in,
                              void* d_out, int out_size, void* d_ws, size_t ws_size,
                              hipStream_t stream) {
    const float* z = (const float*)d_in[0];   // [32768, 256] fp32
    const float* W = (const float*)d_in[1];   // [8192, 256] fp32

    // workspace: Wh (4MB) | Wl (4MB) | wnorm (32KB)
    bf16_t* Wh    = (bf16_t*)d_ws;
    bf16_t* Wl    = Wh + (size_t)KCODES * EDIM;
    float*  wnorm = (float*)(Wl + (size_t)KCODES * EDIM);

    float* out_zq   = (float*)d_out;                       // [N, 256]
    float* out_idx  = out_zq + (size_t)NPTS * EDIM;        // [N]
    float* out_loss = out_idx + NPTS;                      // [N]

    (void)hipFuncSetAttribute((const void*)vq_main,
                              hipFuncAttributeMaxDynamicSharedMemorySize, LDS_TOT);

    vq_prep<<<KCODES, 256, 0, stream>>>(W, Wh, Wl, wnorm);
    vq_main<<<NPTS / MTILE, 256, LDS_TOT, stream>>>(z, W, Wh, Wl, wnorm,
                                                    out_zq, out_idx, out_loss);
}
